// MultiHeadAttention_23012434772155
// MI455X (gfx1250) — compile-verified, matrix operand order checked by reading, not run on hardware
//
#include <hip/hip_runtime.h>

typedef __attribute__((ext_vector_type(8)))  float   v8f;
typedef __attribute__((ext_vector_type(16))) __bf16  v16bf;
typedef __attribute__((ext_vector_type(8)))  __bf16  v8bf;

#define B_SZ 8
#define S_SZ 2048
#define DM   384
#define NH   8
#define DH   48
#define DHP  64                      // padded head dim (K multiple of 32)
#define BH   (B_SZ * NH)             // 64
#define CTX_ELEMS (B_SZ * S_SZ * DM) // 6291456

__device__ __forceinline__ __bf16 f2bf(float f) {
  union { float f; unsigned u; } a; a.f = f;
  unsigned u = a.u + 0x7FFFu + ((a.u >> 16) & 1u);  // round-to-nearest-even
  union { unsigned short s; __bf16 b; } o; o.s = (unsigned short)(u >> 16);
  return o.b;
}

__device__ __forceinline__ v8f vzero8() {
  v8f r;
#pragma unroll
  for (int i = 0; i < 8; ++i) r[i] = 0.0f;
  return r;
}

// Per-lane WMMA operand: two contiguous 8-element bf16 chunks at p and p+16
// (ISA 16-bit A/B layout: lanes<16 hold K 0-7 & 16-23, lanes>=16 hold K 8-15 & 24-31;
//  caller passes p already offset by the lane's K-half.)
__device__ __forceinline__ v16bf ld16_bf(const __bf16* p) {
  v8bf lo = *(const v8bf*)(p);
  v8bf hi = *(const v8bf*)(p + 16);
  v16bf r;
#pragma unroll
  for (int i = 0; i < 8; ++i) { r[i] = lo[i]; r[i + 8] = hi[i]; }
  return r;
}

__device__ __forceinline__ v16bf ld16_f32cvt(const float* p) {
  v8f lo = *(const v8f*)(p);
  v8f hi = *(const v8f*)(p + 16);
  v16bf r;
#pragma unroll
  for (int i = 0; i < 8; ++i) { r[i] = f2bf(lo[i]); r[i + 8] = f2bf(hi[i]); }
  return r;
}

__device__ __forceinline__ v8f wmma_bf16(v16bf a, v16bf b, v8f c) {
  return __builtin_amdgcn_wmma_f32_16x16x32_bf16(false, a, false, b, (short)0, c,
                                                 false, false);
}

// ---------------------------------------------------------------- zero fill
__global__ void zero_fill_kernel(float4* p) {
  const size_t i = (size_t)blockIdx.x * blockDim.x + threadIdx.x;
  float4 z; z.x = z.y = z.z = z.w = 0.0f;
  p[i] = z;
}

// ------------------------------------------------- weight transpose+convert
// Wt[n*DM + k] = bf16(W[k*DM + n]) for each of the 3 weights.
__global__ void wconv_kernel(const float* __restrict__ Wq, const float* __restrict__ Wk,
                             const float* __restrict__ Wv, __bf16* __restrict__ Wt) {
  const int which = blockIdx.y;
  const float* W = (which == 0) ? Wq : (which == 1) ? Wk : Wv;
  __bf16* T = Wt + which * DM * DM;
  const int idx = blockIdx.x * blockDim.x + threadIdx.x;  // < DM*DM
  const int k = idx / DM;
  const int n = idx - k * DM;
  T[n * DM + k] = f2bf(W[idx]);
}

// ------------------------------------------------------ QKV projection GEMM
// out = X(16384x384) @ W(384x384) + bias, via WMMA bf16, one 16x16 tile/wave.
// q,k stored [bh][s][64] (padded bf16); v stored transposed [bh][64][s].
__global__ void __launch_bounds__(128)
proj_kernel(const float* __restrict__ Xq, const float* __restrict__ Xk,
            const float* __restrict__ Xv,
            const float* __restrict__ bq, const float* __restrict__ bk,
            const float* __restrict__ bv,
            const __bf16* __restrict__ Wt,
            __bf16* __restrict__ qws, __bf16* __restrict__ kws,
            __bf16* __restrict__ vT) {
  const int which = blockIdx.z;
  const float* X    = (which == 0) ? Xq : (which == 1) ? Xk : Xv;
  const float* bias = (which == 0) ? bq : (which == 1) ? bk : bv;
  const __bf16* W = Wt + which * DM * DM;

  const int lane  = threadIdx.x & 31;
  const int wave  = threadIdx.x >> 5;
  const int m0    = blockIdx.x * 16;
  const int n0    = (blockIdx.y * 4 + wave) * 16;
  const int lrow  = lane & 15;
  const int khalf = (lane >> 4) << 3;

  const float*  Arow = X + (long)(m0 + lrow) * DM;
  const __bf16* Brow = W + (long)(n0 + lrow) * DM;
  v8f acc = vzero8();
  for (int kk = 0; kk < DM; kk += 32) {
    v16bf A  = ld16_f32cvt(Arow + kk + khalf);
    v16bf Bm = ld16_bf(Brow + kk + khalf);
    acc = wmma_bf16(A, Bm, acc);
  }

  const int   ncol  = n0 + lrow;
  const float bval  = bias[ncol];
  const int   h     = ncol / DH;
  const int   d     = ncol - h * DH;
  const int   rbase = m0 + ((lane >> 4) << 3);
#pragma unroll
  for (int r = 0; r < 8; ++r) {
    const int m = rbase + r;
    const int b = m >> 11;
    const int s = m & (S_SZ - 1);
    const __bf16 val = f2bf(acc[r] + bval);
    const int bh = b * NH + h;
    if (which == 2)      vT [((long)bh * DHP + d) * S_SZ + s] = val;
    else if (which == 0) qws[((long)bh * S_SZ + s) * DHP + d] = val;
    else                 kws[((long)bh * S_SZ + s) * DHP + d] = val;
  }
}

// -------------------------------- fused scores + softmax + attn-write + context
// One block = one (bh, 16-query-row tile). 8 waves. 16x2048 f32 score tile in LDS.
__global__ void __launch_bounds__(256)
attn_kernel(const __bf16* __restrict__ qws, const __bf16* __restrict__ kws,
            const __bf16* __restrict__ vT,
            float* __restrict__ attn, float* __restrict__ ctx) {
  extern __shared__ float smem[];
  float* tile   = smem;                 // 16 * 2048
  float* part   = tile + 16 * S_SZ;     // 8 * 768 (context partials)
  float* red    = part + 8 * 768;       // 256
  float* rowmax = red + 256;            // 16
  float* rowinv = rowmax + 16;          // 16

  const int tid  = threadIdx.x;
  const int lane = tid & 31;
  const int wave = tid >> 5;
  const int bh   = blockIdx.z;
  const int b    = bh >> 3;
  const int h    = bh & 7;
  const int m0   = blockIdx.x * 16;

  const int lrow  = lane & 15;
  const int khalf = (lane >> 4) << 3;
  const __bf16* qb = qws + (long)bh * S_SZ * DHP;
  const __bf16* kb = kws + (long)bh * S_SZ * DHP;
  const __bf16* vb = vT  + (long)bh * DHP * S_SZ;

  // ---- Phase 1: raw scores -> LDS (each wave covers 256 key columns) ----
  {
    const __bf16* aptr = qb + (long)(m0 + lrow) * DHP + khalf;
    const v16bf A0 = ld16_bf(aptr);
    const v16bf A1 = ld16_bf(aptr + 32);
    const float scale = 0.14433756729740643f;  // 1/sqrt(48)
    const int rb2 = (lane >> 4) << 3;
    for (int t = 0; t < 16; ++t) {
      const int n0 = wave * 256 + t * 16;
      const __bf16* bptr = kb + (long)(n0 + lrow) * DHP + khalf;
      v8f acc = vzero8();
      acc = wmma_bf16(A0, ld16_bf(bptr), acc);
      acc = wmma_bf16(A1, ld16_bf(bptr + 32), acc);
#pragma unroll
      for (int r = 0; r < 8; ++r)
        tile[(rb2 + r) * S_SZ + n0 + lrow] = acc[r] * scale;
    }
  }
  __syncthreads();

  // ---- Phase 2: softmax in LDS (tile holds exp(x - rowmax) afterwards) ----
  const int srow = tid >> 4;   // 0..15
  const int g    = tid & 15;   // 0..15
  {
    float mx = -3.0e38f;
    for (int i = 0; i < 128; ++i) mx = fmaxf(mx, tile[srow * S_SZ + g + i * 16]);
    red[tid] = mx;
    __syncthreads();
    if (tid < 16) {
      float m2 = red[tid * 16];
      for (int j = 1; j < 16; ++j) m2 = fmaxf(m2, red[tid * 16 + j]);
      rowmax[tid] = m2;
    }
    __syncthreads();
    const float rm = rowmax[srow];
    float sum = 0.0f;
    for (int i = 0; i < 128; ++i) {
      const int idx = srow * S_SZ + g + i * 16;
      const float e = __expf(tile[idx] - rm);
      tile[idx] = e;
      sum += e;
    }
    red[tid] = sum;
    __syncthreads();
    if (tid < 16) {
      float s2 = 0.0f;
      for (int j = 0; j < 16; ++j) s2 += red[tid * 16 + j];
      rowinv[tid] = 1.0f / s2;
    }
    __syncthreads();
  }

  // ---- Phase 3: single coalesced write of normalized attention ----
  {
    float* arow = attn + ((long)(h * B_SZ + b) * S_SZ + m0) * S_SZ;
    for (int r = 0; r < 16; ++r) {
      const float inv = rowinv[r];
#pragma unroll
      for (int j = 0; j < 8; ++j) {
        const int c = tid + j * 256;
        arow[(long)r * S_SZ + c] = tile[r * S_SZ + c] * inv;
      }
    }
  }

  // ---- Phase 4: context = (e @ V) * inv, A read from LDS, K split 8 ways ----
  {
    v8f acc0 = vzero8(), acc1 = vzero8(), acc2 = vzero8();
    const int kk0 = wave * 256;
    for (int kk = kk0; kk < kk0 + 256; kk += 32) {
      const int kbo = kk + khalf;
      const v16bf A = ld16_f32cvt(tile + lrow * S_SZ + kbo);
      acc0 = wmma_bf16(A, ld16_bf(vb + (long)(lrow) * S_SZ + kbo), acc0);
      acc1 = wmma_bf16(A, ld16_bf(vb + (long)(16 + lrow) * S_SZ + kbo), acc1);
      acc2 = wmma_bf16(A, ld16_bf(vb + (long)(32 + lrow) * S_SZ + kbo), acc2);
    }
    const int rb2 = (lane >> 4) << 3;
#pragma unroll
    for (int r = 0; r < 8; ++r) {
      part[wave * 768 + (rb2 + r) * 48 + lrow]      = acc0[r];
      part[wave * 768 + (rb2 + r) * 48 + 16 + lrow] = acc1[r];
      part[wave * 768 + (rb2 + r) * 48 + 32 + lrow] = acc2[r];
    }
  }
  __syncthreads();
  for (int e = tid; e < 768; e += 256) {
    float s2 = 0.0f;
#pragma unroll
    for (int w2 = 0; w2 < 8; ++w2) s2 += part[w2 * 768 + e];
    const int r = e / 48;
    const int c = e - r * 48;
    ctx[((long)(b * S_SZ + m0 + r)) * DM + h * DH + c] = s2 * rowinv[r];
  }
}

// --------------------------------------------------------------------------
extern "C" void kernel_launch(void* const* d_in, const int* in_sizes, int n_in,
                              void* d_out, int out_size, void* d_ws, size_t ws_size,
                              hipStream_t stream) {
  (void)in_sizes; (void)n_in; (void)out_size; (void)ws_size;
  const float* q  = (const float*)d_in[0];
  const float* k  = (const float*)d_in[1];
  const float* v  = (const float*)d_in[2];
  const float* Wq = (const float*)d_in[3];
  const float* bq = (const float*)d_in[4];
  const float* Wk = (const float*)d_in[5];
  const float* bk = (const float*)d_in[6];
  const float* Wv = (const float*)d_in[7];
  const float* bv = (const float*)d_in[8];
  float* out  = (float*)d_out;
  float* attn = out + (size_t)CTX_ELEMS;

  char* ws = (char*)d_ws;
  const size_t QKV_BYTES = (size_t)BH * S_SZ * DHP * sizeof(__bf16);  // 16 MiB each
  __bf16* qws = (__bf16*)(ws);
  __bf16* kws = (__bf16*)(ws + QKV_BYTES);
  __bf16* vT  = (__bf16*)(ws + 2 * QKV_BYTES);
  __bf16* Wt  = (__bf16*)(ws + 3 * QKV_BYTES);  // 3 * 384*384 bf16

  // Zero padded q/k/vT buffers (pad lanes d=48..63 must be 0 for the K=64 WMMA loop)
  zero_fill_kernel<<<dim3((unsigned)(3 * QKV_BYTES / (16 * 256))), 256, 0, stream>>>(
      (float4*)ws);
  wconv_kernel<<<dim3((DM * DM) / 256, 3), 256, 0, stream>>>(Wq, Wk, Wv, Wt);
  proj_kernel<<<dim3(B_SZ * S_SZ / 16, (DM / 16) / 4, 3), 128, 0, stream>>>(
      q, k, v, bq, bk, bv, Wt, qws, kws, vT);
  const size_t smem = (size_t)(16 * S_SZ + 8 * 768 + 256 + 32) * sizeof(float);
  attn_kernel<<<dim3(S_SZ / 16, 1, BH), 256, smem, stream>>>(qws, kws, vT, attn, out);
}
